// GRUEncoder_27745488732303
// MI455X (gfx1250) — matrix operand order repaired and checked by reading, not compiled, not run on hardware
//
#include <hip/hip_runtime.h>

// ---------------------------------------------------------------------------
// GRU encoder for MI455X (gfx1250): 2-layer GRU, B=128, T=512, E=512, H=1024.
// Batch-partitioned persistent workgroups (16 batch rows = one WMMA M-tile per
// block), h-state resident in LDS as bf16, bf16 weights resident in L2,
// v_wmma_f32_16x16x32_bf16 for all projections, both layers fused per step.
// K loop split at the x/h boundary so region select AND the n-gate x/h
// accumulator split are compile-time (no branches around WMMA).
// ---------------------------------------------------------------------------

typedef __attribute__((ext_vector_type(16))) __bf16 v16bf;
typedef __attribute__((ext_vector_type(8)))  __bf16 v8bf;
typedef __attribute__((ext_vector_type(8)))  float  v8f;

constexpr int Bsz = 128, Tsz = 512, Esz = 512, Hsz = 1024;
constexpr int XSTR = Esz + 8;    // LDS row stride (bf16 elems) for x_t staging
constexpr int HSTR = Hsz + 8;    // LDS row stride for hidden state

// ---- A-fragment (16x32 bf16, MxK) from LDS in contiguous k order (B uses the same):
// lanes 0-15: row=lane, K = k0..k0+15 (contiguous)
// lanes 16-31: row=lane-16, K = k0+16..k0+31.
__device__ __forceinline__ v16bf load_a_frag(const __bf16* base, int stride,
                                             int k0, int lane) {
  const int r = lane & 15, half = lane >> 4;
  const __bf16* p = base + r * stride + k0 + half * 16;
  v8bf lo = *(const v8bf*)p;          // 16B ds_load_b128
  v8bf hi = *(const v8bf*)(p + 8);    // 16B ds_load_b128
  v16bf a;
#pragma unroll
  for (int i = 0; i < 8; ++i) { a[i] = lo[i]; a[i + 8] = hi[i]; }
  return a;
}

// ---- B-fragment (32x16 bf16, KxN). W stored as [N][Ktot] (K-contiguous per
// output column), so each lane reads 16 contiguous bf16 = 32 bytes:
// lanes 0-15: col=lane, K=k0..k0+15 ; lanes 16-31: col=lane-16, K=k0+16..k0+31.
__device__ __forceinline__ v16bf load_b_frag(const __bf16* __restrict__ W,
                                             int Ktot, int k0, int ntile,
                                             int lane) {
  const int n = ntile * 16 + (lane & 15);
  const __bf16* p = W + (size_t)n * Ktot + k0 + (lane >> 4) * 16;
  return *(const v16bf*)p;  // 32B contiguous, 32B aligned -> 2x global_load_b128
}

__device__ __forceinline__ float sigm(float x) {
  return 1.f / (1.f + __expf(-x));
}
__device__ __forceinline__ float tanh_f(float x) {
  x = fminf(fmaxf(x, -15.f), 15.f);
  float e = __expf(-2.f * x);
  return (1.f - e) / (1.f + e);
}

#define WMMA_BF16(A, Bm, C) \
  __builtin_amdgcn_wmma_f32_16x16x32_bf16(false, (A), false, (Bm), (short)0, (C), false, false)

// One GRU layer step for this block's 16 batch rows.
// A operand = [regA (KA cols) | regH (Hsz cols)], W = [N=3H][Ktot=KA+H] bf16.
// SPLIT = KA/32 (x/h region boundary == n-gate x/h accumulator split).
// Wave `wid` owns hidden-column tiles jt = wid*2+{0,1}; gate ntiles are
// r: jt, z: 64+jt, n: 128+jt. One A fragment feeds 6 WMMAs per k-step.
template <int KB, int SPLIT, int KA>
__device__ __forceinline__ void gru_layer_step(
    const __bf16* __restrict__ regA, int strideA, __bf16* regH,
    const __bf16* __restrict__ W, int Ktot, const float* __restrict__ bih,
    const float* __restrict__ bhh, int wid, int lane) {
  v8f acc[2][4] = {};
  const int jt0 = wid * 2;

  // Phase 1: K in [0, KA): A comes from regA; n-gate -> accNX.
#pragma unroll 1
  for (int kb = 0; kb < SPLIT; ++kb) {
    const int k0 = kb * 32;
    v16bf a = load_a_frag(regA, strideA, k0, lane);
#pragma unroll
    for (int q = 0; q < 2; ++q) {
      const int jt = jt0 + q;
      v16bf bR = load_b_frag(W, Ktot, k0, jt, lane);
      acc[q][0] = WMMA_BF16(a, bR, acc[q][0]);
      v16bf bZ = load_b_frag(W, Ktot, k0, 64 + jt, lane);
      acc[q][1] = WMMA_BF16(a, bZ, acc[q][1]);
      v16bf bN = load_b_frag(W, Ktot, k0, 128 + jt, lane);
      acc[q][2] = WMMA_BF16(a, bN, acc[q][2]);
    }
  }
  // Phase 2: K in [KA, Ktot): A comes from regH (h state); n-gate -> accNH.
#pragma unroll 1
  for (int kb = SPLIT; kb < KB; ++kb) {
    const int k0 = kb * 32;
    v16bf a = load_a_frag(regH, HSTR, k0 - KA, lane);
#pragma unroll
    for (int q = 0; q < 2; ++q) {
      const int jt = jt0 + q;
      v16bf bR = load_b_frag(W, Ktot, k0, jt, lane);
      acc[q][0] = WMMA_BF16(a, bR, acc[q][0]);
      v16bf bZ = load_b_frag(W, Ktot, k0, 64 + jt, lane);
      acc[q][1] = WMMA_BF16(a, bZ, acc[q][1]);
      v16bf bN = load_b_frag(W, Ktot, k0, 128 + jt, lane);
      acc[q][3] = WMMA_BF16(a, bN, acc[q][3]);
    }
  }

  __syncthreads();  // all waves finished reading regA/regH
#pragma unroll
  for (int q = 0; q < 2; ++q) {
    const int jt = jt0 + q;
    const int j = jt * 16 + (lane & 15);
    const int mb = (lane >> 4) * 8;  // C layout: lanes 16-31 hold rows M=8..15
    const float br  = bih[j] + bhh[j];
    const float bz  = bih[Hsz + j] + bhh[Hsz + j];
    const float bnx = bih[2 * Hsz + j];
    const float bnh = bhh[2 * Hsz + j];
#pragma unroll
    for (int i = 0; i < 8; ++i) {
      const int m = mb + i;
      float r = sigm(acc[q][0][i] + br);
      float z = sigm(acc[q][1][i] + bz);
      float n = tanh_f(acc[q][2][i] + bnx + r * (acc[q][3][i] + bnh));
      float hold = (float)regH[m * HSTR + j];
      regH[m * HSTR + j] = (__bf16)((1.f - z) * n + z * hold);  // lane-owned
    }
  }
  __syncthreads();  // h_new visible to everyone
}

__global__ void __launch_bounds__(1024) gru_encoder_kernel(
    const int* __restrict__ src, const float* __restrict__ emb,
    const float* __restrict__ bih0, const float* __restrict__ bhh0,
    const float* __restrict__ bih1, const float* __restrict__ bhh1,
    const __bf16* __restrict__ W0, const __bf16* __restrict__ W1,
    float* __restrict__ out) {
  __shared__ alignas(64) __bf16 sXA[16 * XSTR];  // x_t (gathered embedding rows)
  __shared__ alignas(64) __bf16 sH0[16 * HSTR];  // layer-0 hidden state
  __shared__ alignas(64) __bf16 sH1[16 * HSTR];  // layer-1 hidden state

  const int tid = threadIdx.x;
  const int blk = blockIdx.x;      // owns batch rows [16*blk, 16*blk+16)
  const int lane = tid & 31;
  const int wid = tid >> 5;        // 0..31 (wave32)

  for (int i = tid; i < 16 * HSTR; i += 1024) {
    sH0[i] = (__bf16)0.f;
    sH1[i] = (__bf16)0.f;
  }
  __syncthreads();

#pragma unroll 1
  for (int t = 0; t < Tsz; ++t) {
    {  // gather x_t: 16 rows x 512 cols, 8 elems/thread, f32 -> bf16
      const int row = tid >> 6;
      const int cb = (tid & 63) * 8;
      const int b = blk * 16 + row;
      const int v = src[b * Tsz + t];
      const float4* p = (const float4*)(emb + (size_t)v * Esz + cb);
      float4 f0 = p[0], f1 = p[1];
      __bf16* dst = sXA + row * XSTR + cb;
      dst[0] = (__bf16)f0.x; dst[1] = (__bf16)f0.y;
      dst[2] = (__bf16)f0.z; dst[3] = (__bf16)f0.w;
      dst[4] = (__bf16)f1.x; dst[5] = (__bf16)f1.y;
      dst[6] = (__bf16)f1.z; dst[7] = (__bf16)f1.w;
    }
    __syncthreads();
    // layer 0: A = [x_t | h0], K = 512+1024 = 1536
    gru_layer_step<48, 16, 512>(sXA, XSTR, sH0, W0, Esz + Hsz, bih0, bhh0, wid, lane);
    // layer 1: A = [h0_new | h1], K = 1024+1024 = 2048
    gru_layer_step<64, 32, 1024>(sH0, HSTR, sH1, W1, 2 * Hsz, bih1, bhh1, wid, lane);
  }

  {  // emit final hidden states: out[layer][b][j], f32
    const int row = tid >> 6;
    const int cb = (tid & 63) * 16;
    const int b = blk * 16 + row;
#pragma unroll 1
    for (int c = 0; c < 16; ++c) {
      out[(size_t)b * Hsz + cb + c] = (float)sH0[row * HSTR + cb + c];
      out[(size_t)Bsz * Hsz + (size_t)b * Hsz + cb + c] =
          (float)sH1[row * HSTR + cb + c];
    }
  }
}

// Build K-contiguous bf16 weight matrix  Wcat[n][k] = k<Ka ? wa[n][k] : wb[n][k-Ka]
__global__ void concat_cvt_bf16(const float* __restrict__ wa,
                                const float* __restrict__ wb,
                                __bf16* __restrict__ dst, int Ka, int Kb,
                                long total) {
  long i = (long)blockIdx.x * blockDim.x + threadIdx.x;
  if (i >= total) return;
  const int Kt = Ka + Kb;
  const long n = i / Kt;
  const int k = (int)(i % Kt);
  float v = (k < Ka) ? wa[n * Ka + k] : wb[n * Kb + (k - Ka)];
  dst[i] = (__bf16)v;
}

extern "C" void kernel_launch(void* const* d_in, const int* in_sizes, int n_in,
                              void* d_out, int out_size, void* d_ws,
                              size_t ws_size, hipStream_t stream) {
  const int*   src  = (const int*)d_in[0];
  const float* emb  = (const float*)d_in[1];
  const float* wih0 = (const float*)d_in[2];
  const float* whh0 = (const float*)d_in[3];
  const float* bih0 = (const float*)d_in[4];
  const float* bhh0 = (const float*)d_in[5];
  const float* wih1 = (const float*)d_in[6];
  const float* whh1 = (const float*)d_in[7];
  const float* bih1 = (const float*)d_in[8];
  const float* bhh1 = (const float*)d_in[9];
  float* out = (float*)d_out;

  // workspace: W0cat [3H][E+H] bf16 (9.4 MB), W1cat [3H][2H] bf16 (12.6 MB)
  __bf16* W0 = (__bf16*)d_ws;
  const size_t w0_elems = (size_t)3 * Hsz * (Esz + Hsz);
  __bf16* W1 = (__bf16*)((char*)d_ws + w0_elems * sizeof(__bf16));
  const size_t w1_elems = (size_t)3 * Hsz * (2 * Hsz);

  concat_cvt_bf16<<<(int)((w0_elems + 255) / 256), 256, 0, stream>>>(
      wih0, whh0, W0, Esz, Hsz, (long)w0_elems);
  concat_cvt_bf16<<<(int)((w1_elems + 255) / 256), 256, 0, stream>>>(
      wih1, whh1, W1, Hsz, Hsz, (long)w1_elems);

  gru_encoder_kernel<<<Bsz / 16, 1024, 0, stream>>>(
      src, emb, bih0, bhh0, bih1, bhh1, W0, W1, out);
}
